// RelationalTransformer_6717328851660
// MI455X (gfx1250) — matrix operand order repaired and checked by reading, not compiled, not run on hardware
//
#include <hip/hip_runtime.h>
#include <math.h>

// ---------------------------------------------------------------------------
// Types / helpers
// ---------------------------------------------------------------------------
typedef __attribute__((ext_vector_type(16))) __bf16 v16bf;
typedef __attribute__((ext_vector_type(8)))  float  v8f;
typedef __attribute__((ext_vector_type(4)))  int    v4i_;
typedef unsigned short bf16raw;

union Frag16 { v16bf v; uint4 q[2]; };

__device__ __forceinline__ bf16raw f2bf(float f) {
    union { float f; unsigned u; } x; x.f = f;
    unsigned r = x.u + 0x7FFFu + ((x.u >> 16) & 1u);   // round-to-nearest-even
    return (bf16raw)(r >> 16);
}
__device__ __forceinline__ float bf2f(bf16raw h) {
    union { unsigned u; float f; } x; x.u = ((unsigned)h) << 16; return x.f;
}
__device__ __forceinline__ float gelu_exact(float x) {
    return 0.5f * x * (1.0f + erff(x * 0.70710678118654752f));
}

// ---- CDNA5 async global->LDS path (ASYNCcnt-tracked, bypasses VGPRs) ------
#ifndef __has_builtin
#define __has_builtin(x) 0
#endif
#if __has_builtin(__builtin_amdgcn_global_load_async_to_lds_b128) && \
    __has_builtin(__builtin_amdgcn_s_wait_asynccnt)
#define HAVE_ASYNC_LDS 1
#else
#define HAVE_ASYNC_LDS 0
#endif

__device__ __forceinline__ void async_b128(const void* g, void* l) {
#if HAVE_ASYNC_LDS
    __builtin_amdgcn_global_load_async_to_lds_b128(
        (__attribute__((address_space(1))) v4i_*)(g),
        (__attribute__((address_space(3))) v4i_*)(l), 0, 0);
#else
    *(uint4*)l = *(const uint4*)g;
#endif
}
__device__ __forceinline__ void wait_async_lds() {
#if HAVE_ASYNC_LDS
    __builtin_amdgcn_s_wait_asynccnt(0);
#endif
}

// ---------------------------------------------------------------------------
// bf16 WMMA GEMM:  C[M,N] = act( A[M,K] @ W[K,N] + bias + s_i + t_j )
//   A bf16 row-major [M,K];  Wt bf16 = W TRANSPOSED, row-major [N,K].
//   flags: bit0 = +bias[N], bit1 = exact GELU,
//          bit2 = edge broadcast add: row=(b,i,j) -> += Sv[b,i,:] + Tv[b,j,:]
// Block tile 64x128x32, 128 threads (4 waves), wave tile 32x64 = 2x4 WMMA.
// Double-buffered LDS with async copies overlapping WMMA.
// ---------------------------------------------------------------------------
#define TILE_M 64
#define TILE_N 128
#define TILE_K 32

__global__ __launch_bounds__(128)
void gemm_bf16_wmma(const bf16raw* __restrict__ A, const bf16raw* __restrict__ Wt,
                    float* __restrict__ Cf, bf16raw* __restrict__ Cb,
                    const float* __restrict__ bias,
                    const float* __restrict__ Sv, const float* __restrict__ Tv,
                    int M, int N, int K, int flags)
{
    __shared__ __align__(16) bf16raw As[2][TILE_M][TILE_K];   // [m][k]
    __shared__ __align__(16) bf16raw Bs[2][TILE_N][TILE_K];   // [n][k]

    const int tid   = threadIdx.x;
    const int lane  = tid & 31;
    const int wid   = tid >> 5;
    const int waveM = wid >> 1;          // 0..1 -> 32 rows each
    const int waveN = wid & 1;           // 0..1 -> 64 cols each
    const int tileM = blockIdx.y * TILE_M;
    const int tileN = blockIdx.x * TILE_N;

    const int  l  = lane & 15;
    const bool hi = (lane >= 16);

    v8f acc[2][4];
    #pragma unroll
    for (int mt = 0; mt < 2; ++mt)
        #pragma unroll
        for (int nt = 0; nt < 4; ++nt)
            #pragma unroll
            for (int r = 0; r < 8; ++r)
                acc[mt][nt][r] = 0.0f;

    auto stage = [&](int buf, int k0) {
        // A tile: 64x32 = 256 16B chunks, 2 per thread
        #pragma unroll
        for (int c = 0; c < 2; ++c) {
            int chunk = tid + c * 128;
            int row   = chunk >> 2;
            int kc    = (chunk & 3) << 3;
            async_b128(A + (size_t)(tileM + row) * K + k0 + kc, &As[buf][row][kc]);
        }
        // B tile (pre-transposed weights): 128x32 = 512 chunks, 4 per thread
        #pragma unroll
        for (int c = 0; c < 4; ++c) {
            int chunk = tid + c * 128;
            int n     = chunk >> 2;
            int kc    = (chunk & 3) << 3;
            async_b128(Wt + (size_t)(tileN + n) * K + k0 + kc, &Bs[buf][n][kc]);
        }
#if !HAVE_ASYNC_LDS
        if (k0 + TILE_K < K)
            __builtin_prefetch(A + (size_t)(tileM + (tid >> 2)) * K + k0 + TILE_K, 0, 1);
#endif
    };

    const int nk = K / TILE_K;
    stage(0, 0);
    for (int t = 0; t < nk; ++t) {
        const int cur = t & 1;
        wait_async_lds();          // this wave's copies done
        __syncthreads();           // whole tile visible to all waves
        if (t + 1 < nk) stage(cur ^ 1, (t + 1) * TILE_K);

        Frag16 a[2], b[4];
        #pragma unroll
        for (int mt = 0; mt < 2; ++mt) {
            int m  = waveM * 32 + mt * 16 + l;
            int ko = hi ? 8 : 0;           // A: lo lanes K0..7/K16..23, hi K8..15/K24..31
            a[mt].q[0] = *(const uint4*)&As[cur][m][ko];
            a[mt].q[1] = *(const uint4*)&As[cur][m][16 + ko];
        }
        #pragma unroll
        for (int nt = 0; nt < 4; ++nt) {
            int n  = waveN * 64 + nt * 16 + l;
            int ko = hi ? 8 : 0;           // B: lo lanes K0..7/K16..23, hi K8..15/K24..31
            b[nt].q[0] = *(const uint4*)&Bs[cur][n][ko];
            b[nt].q[1] = *(const uint4*)&Bs[cur][n][16 + ko];
        }
        #pragma unroll
        for (int mt = 0; mt < 2; ++mt)
            #pragma unroll
            for (int nt = 0; nt < 4; ++nt)
                acc[mt][nt] = __builtin_amdgcn_wmma_f32_16x16x32_bf16(
                    false, a[mt].v, false, b[nt].v,
                    (short)0, acc[mt][nt], false, false);
    }

    // epilogue: C/D layout -> lanes 0-15: M=r, hi lanes: M=r+8; N = lane&15
    #pragma unroll
    for (int mt = 0; mt < 2; ++mt) {
        #pragma unroll
        for (int nt = 0; nt < 4; ++nt) {
            #pragma unroll
            for (int r = 0; r < 8; ++r) {
                int row = tileM + waveM * 32 + mt * 16 + (hi ? r + 8 : r);
                int col = tileN + waveN * 64 + nt * 16 + l;
                float v = acc[mt][nt][r];
                if (flags & 1) v += bias[col];
                if (flags & 4) {
                    int bb = row >> 14, ii = (row >> 7) & 127, jj = row & 127;
                    v += Sv[(size_t)((bb << 7) + ii) * N + col]
                       + Tv[(size_t)((bb << 7) + jj) * N + col];
                }
                if (flags & 2) v = gelu_exact(v);
                size_t o = (size_t)row * N + col;
                if (Cf) Cf[o] = v;
                if (Cb) Cb[o] = f2bf(v);
            }
        }
    }
}

// ---------------------------------------------------------------------------
// Fused edge-modulated attention. One block per (b,h,i); 128 threads.
// ---------------------------------------------------------------------------
__global__ __launch_bounds__(128)
void attn_fused(const float* __restrict__ qkvn, const bf16raw* __restrict__ qkve,
                float* __restrict__ attn, float* __restrict__ outv)
{
    const int idx = blockIdx.x;           // b*1024 + h*128 + i
    const int b = idx >> 10;
    const int h = (idx >> 7) & 7;
    const int i = idx & 127;
    const int j = threadIdx.x;

    __shared__ float qsh[32];
    __shared__ float red[128];
    __shared__ float ash[128];
    __shared__ float psum[4][32];

    const float* qn = qkvn + (size_t)(b * 128 + i) * 768 + h * 96;
    if (j < 32) qsh[j] = qn[j];
    __syncthreads();

    const float*   knj = qkvn + (size_t)(b * 128 + j) * 768 + h * 96 + 32;
    const bf16raw* ep  = qkve + ((size_t)(b * 128 + i) * 128 + j) * 1024 + h * 128;

    float dot = 0.0f;
    #pragma unroll 4
    for (int d = 0; d < 32; ++d) {
        float q = qsh[d] + bf2f(ep[d]);
        float k = knj[d] + bf2f(ep[32 + d]);
        dot += q * k;
    }
    dot *= 0.0625f;                        // 1/sqrt(D_HID=256)

    red[j] = dot; __syncthreads();
    for (int s = 64; s > 0; s >>= 1) { if (j < s) red[j] = fmaxf(red[j], red[j + s]); __syncthreads(); }
    float mx = red[0]; __syncthreads();
    float e = __expf(dot - mx);
    red[j] = e; __syncthreads();
    for (int s = 64; s > 0; s >>= 1) { if (j < s) red[j] += red[j + s]; __syncthreads(); }
    float a = e / red[0];

    ash[j] = a;
    attn[(size_t)((b * 8 + h) * 128 + i) * 128 + j] = a;
    __syncthreads();

    // out[b,i,h,:] = sum_j a_j * (v_n*me + ve); 4 j-groups x 32 d lanes
    const int d2 = j & 31, grp = j >> 5;
    float acc = 0.0f;
    for (int jj = grp * 32; jj < grp * 32 + 32; ++jj) {
        float vn = qkvn[(size_t)(b * 128 + jj) * 768 + h * 96 + 64 + d2];
        const bf16raw* ej = qkve + ((size_t)(b * 128 + i) * 128 + jj) * 1024 + h * 128;
        acc += ash[jj] * (vn * bf2f(ej[96 + d2]) + bf2f(ej[64 + d2]));
    }
    psum[grp][d2] = acc; __syncthreads();
    if (j < 32)
        outv[(size_t)(b * 128 + i) * 256 + h * 32 + j] =
            psum[0][j] + psum[1][j] + psum[2][j] + psum[3][j];
}

// ---------------------------------------------------------------------------
// LayerNorm with residual:  y = LN(resid + delta) * g + b ;  blockDim == D
// ---------------------------------------------------------------------------
__global__ void ln_kernel(const float* __restrict__ resid, const float* __restrict__ delta,
                          const float* __restrict__ g, const float* __restrict__ bta,
                          float* __restrict__ outf, bf16raw* __restrict__ outb, int D)
{
    const int row = blockIdx.x;
    const int t   = threadIdx.x;
    __shared__ float red[256];
    const size_t off = (size_t)row * D + t;
    float x = resid[off] + delta[off];
    red[t] = x; __syncthreads();
    for (int s = D >> 1; s > 0; s >>= 1) { if (t < s) red[t] += red[t + s]; __syncthreads(); }
    float mean = red[0] / (float)D; __syncthreads();
    float dv = x - mean;
    red[t] = dv * dv; __syncthreads();
    for (int s = D >> 1; s > 0; s >>= 1) { if (t < s) red[t] += red[t + s]; __syncthreads(); }
    float var = red[0] / (float)D;
    float y = dv * rsqrtf(var + 1e-5f) * g[t] + bta[t];
    if (outf) outf[off] = y;
    if (outb) outb[off] = f2bf(y);
}

// ---------------------------------------------------------------------------
// f32 -> bf16 conversion (activations)
// ---------------------------------------------------------------------------
__global__ void cvt_f32_bf16(const float* __restrict__ in, bf16raw* __restrict__ out, int n)
{
    int i = blockIdx.x * blockDim.x + threadIdx.x;
    if (i < n) out[i] = f2bf(in[i]);
}

// ---------------------------------------------------------------------------
// Weight transpose+convert: Wt[n*K+k] = bf16(W[k*N+n])   (writes coalesced)
// ---------------------------------------------------------------------------
__global__ void wtrans_bf16(const float* __restrict__ W, bf16raw* __restrict__ Wt, int K, int N)
{
    int idx = blockIdx.x * blockDim.x + threadIdx.x;
    if (idx >= K * N) return;
    int n = idx / K, k = idx - n * K;
    Wt[idx] = f2bf(W[(size_t)k * N + n]);
}

// ---------------------------------------------------------------------------
// e_cat = [edge, edge^T] along channels (bf16). total = B*N*N*256.
// ---------------------------------------------------------------------------
__global__ void ecat_kernel(const float* __restrict__ edge, bf16raw* __restrict__ out, int total)
{
    int idx = blockIdx.x * blockDim.x + threadIdx.x;
    if (idx >= total) return;
    int c = idx & 255;
    int m = idx >> 8;                       // row = ((b*128)+i)*128+j
    int jj = m & 127, ii = (m >> 7) & 127, bb = m >> 14;
    float v;
    if (c < 128) v = edge[(size_t)m * 128 + c];
    else         v = edge[((size_t)(bb << 14) + (jj << 7) + ii) * 128 + (c - 128)];
    out[idx] = f2bf(v);
}

// ---------------------------------------------------------------------------
// Host orchestration
// ---------------------------------------------------------------------------
extern "C" void kernel_launch(void* const* d_in, const int* in_sizes, int n_in,
                              void* d_out, int out_size, void* d_ws, size_t ws_size,
                              hipStream_t stream)
{
    (void)in_sizes; (void)n_in; (void)out_size; (void)ws_size;

    const float* node_in = (const float*)d_in[0];
    const float* edge_in = (const float*)d_in[1];
    /* d_in[2] = mask (unused by the math) */
    const float* w_qkv_node = (const float*)d_in[3];
    const float* w_qkv_edge = (const float*)d_in[4];
    const float* w_proj = (const float*)d_in[5];  const float* b_proj = (const float*)d_in[6];
    const float* w_lin0 = (const float*)d_in[7];  const float* b_lin0 = (const float*)d_in[8];
    const float* ln0_g  = (const float*)d_in[9];  const float* ln0_b  = (const float*)d_in[10];
    const float* w_nm1  = (const float*)d_in[11];
    const float* w_nm2  = (const float*)d_in[12]; const float* b_nm2  = (const float*)d_in[13];
    const float* ln1_g  = (const float*)d_in[14]; const float* ln1_b  = (const float*)d_in[15];
    const float* w_e0e  = (const float*)d_in[16]; const float* b_e0e  = (const float*)d_in[17];
    const float* w_e0s  = (const float*)d_in[18]; const float* b_e0s  = (const float*)d_in[19];
    const float* w_e0t  = (const float*)d_in[20]; const float* b_e0t  = (const float*)d_in[21];
    const float* w_e1   = (const float*)d_in[22]; const float* b_e1   = (const float*)d_in[23];
    const float* eln0_g = (const float*)d_in[24]; const float* eln0_b = (const float*)d_in[25];
    const float* w_em1  = (const float*)d_in[26];
    const float* w_em2  = (const float*)d_in[27]; const float* b_em2  = (const float*)d_in[28];
    const float* eln1_g = (const float*)d_in[29]; const float* eln1_b = (const float*)d_in[30];

    const size_t MN = 512;          // B*N
    const size_t ME = 65536;        // B*N*N

    float* out_node = (float*)d_out;                      // [512,256]
    float* out_edge = out_node + MN * 256;                // [65536,128]
    float* out_attn = out_edge + ME * 128;                // [4,8,128,128]

    char* wsp = (char*)d_ws; size_t off = 0;
    auto alloc = [&](size_t bytes) -> void* {
        void* p = wsp + off; off += (bytes + 255) & ~(size_t)255; return p;
    };
    auto af = [&](size_t n) -> float*   { return (float*)alloc(n * 4); };
    auto ab = [&](size_t n) -> bf16raw* { return (bf16raw*)alloc(n * 2); };

    // f32 intermediates
    float* qkvn_f  = af(MN * 768);
    float* outv_f  = af(MN * 256);
    float* lin0_f  = af(MN * 256);
    float* node1_f = af(MN * 256);
    float* nm2_f   = af(MN * 256);
    float* s_f     = af(MN * 512);
    float* t_f     = af(MN * 512);
    float* eout_f  = af(ME * 128);
    float* edge1_f = af(ME * 128);
    float* em2_f   = af(ME * 128);
    // bf16 activations
    bf16raw* node_bf   = ab(MN * 256);
    bf16raw* edge_bf   = ab(ME * 128);
    bf16raw* qkve_bf   = ab(ME * 1024);
    bf16raw* outv_bf   = ab(MN * 256);
    bf16raw* aproj_bf  = ab(MN * 256);
    bf16raw* node1_bf  = ab(MN * 256);
    bf16raw* node2_bf  = ab(MN * 256);
    bf16raw* h_bf      = ab(MN * 1024);
    bf16raw* ecat_bf   = ab(ME * 256);
    bf16raw* ehid_bf   = ab(ME * 512);
    bf16raw* edge1_bf  = ab(ME * 128);
    bf16raw* eh2_bf    = ab(ME * 512);
    // bf16 transposed weights Wt[N][K]
    bf16raw* wqn_t  = ab(256 * 768);
    bf16raw* wqe_t  = ab(128 * 1024);
    bf16raw* wpr_t  = ab(256 * 256);
    bf16raw* wl0_t  = ab(256 * 256);
    bf16raw* wn1_t  = ab(256 * 1024);
    bf16raw* wn2_t  = ab(1024 * 256);
    bf16raw* we0e_t = ab(256 * 512);
    bf16raw* we0s_t = ab(256 * 512);
    bf16raw* we0t_t = ab(256 * 512);
    bf16raw* we1_t  = ab(512 * 128);
    bf16raw* wm1_t  = ab(128 * 512);
    bf16raw* wm2_t  = ab(512 * 128);

    auto cvt = [&](const float* in, bf16raw* outp, size_t n) {
        cvt_f32_bf16<<<(unsigned)((n + 255) / 256), 256, 0, stream>>>(in, outp, (int)n);
    };
    auto wtr = [&](const float* Wp, bf16raw* Wtp, int K, int N) {
        wtrans_bf16<<<(unsigned)(((size_t)K * N + 255) / 256), 256, 0, stream>>>(Wp, Wtp, K, N);
    };
    auto gemm = [&](const bf16raw* A, const bf16raw* Wtp, float* Cf, bf16raw* Cb,
                    const float* bias, const float* Sv, const float* Tv,
                    int M, int Nn, int K, int flags) {
        dim3 g((unsigned)(Nn / TILE_N), (unsigned)(M / TILE_M));
        gemm_bf16_wmma<<<g, 128, 0, stream>>>(A, Wtp, Cf, Cb, bias, Sv, Tv, M, Nn, K, flags);
    };

    // ---- precision staging -------------------------------------------------
    cvt(node_in, node_bf, MN * 256);
    cvt(edge_in, edge_bf, ME * 128);
    wtr(w_qkv_node, wqn_t, 256, 768);
    wtr(w_qkv_edge, wqe_t, 128, 1024);
    wtr(w_proj, wpr_t, 256, 256);
    wtr(w_lin0, wl0_t, 256, 256);
    wtr(w_nm1,  wn1_t, 256, 1024);
    wtr(w_nm2,  wn2_t, 1024, 256);
    wtr(w_e0e,  we0e_t, 256, 512);
    wtr(w_e0s,  we0s_t, 256, 512);
    wtr(w_e0t,  we0t_t, 256, 512);
    wtr(w_e1,   we1_t, 512, 128);
    wtr(w_em1,  wm1_t, 128, 512);
    wtr(w_em2,  wm2_t, 512, 128);

    // ---- attention ---------------------------------------------------------
    gemm(node_bf, wqn_t, qkvn_f, nullptr, nullptr, nullptr, nullptr, 512, 768, 256, 0);
    gemm(edge_bf, wqe_t, nullptr, qkve_bf, nullptr, nullptr, nullptr, 65536, 1024, 128, 0);
    attn_fused<<<4 * 8 * 128, 128, 0, stream>>>(qkvn_f, qkve_bf, out_attn, outv_f);
    cvt(outv_f, outv_bf, MN * 256);
    gemm(outv_bf, wpr_t, nullptr, aproj_bf, b_proj, nullptr, nullptr, 512, 256, 256, 1);

    // ---- node updates ------------------------------------------------------
    gemm(aproj_bf, wl0_t, lin0_f, nullptr, b_lin0, nullptr, nullptr, 512, 256, 256, 1);
    ln_kernel<<<512, 256, 0, stream>>>(node_in, lin0_f, ln0_g, ln0_b, node1_f, node1_bf, 256);
    gemm(node1_bf, wn1_t, nullptr, h_bf, nullptr, nullptr, nullptr, 512, 1024, 256, 2);
    gemm(h_bf, wn2_t, nm2_f, nullptr, b_nm2, nullptr, nullptr, 512, 256, 1024, 1);
    ln_kernel<<<512, 256, 0, stream>>>(node1_f, nm2_f, ln1_g, ln1_b, out_node, node2_bf, 256);

    // ---- edge updates ------------------------------------------------------
    gemm(node2_bf, we0s_t, s_f, nullptr, b_e0s, nullptr, nullptr, 512, 512, 256, 1);
    gemm(node2_bf, we0t_t, t_f, nullptr, b_e0t, nullptr, nullptr, 512, 512, 256, 1);
    ecat_kernel<<<(unsigned)((ME * 256 + 255) / 256), 256, 0, stream>>>(edge_in, ecat_bf, (int)(ME * 256));
    gemm(ecat_bf, we0e_t, nullptr, ehid_bf, b_e0e, s_f, t_f, 65536, 512, 256, 1 | 2 | 4);
    gemm(ehid_bf, we1_t, eout_f, nullptr, b_e1, nullptr, nullptr, 65536, 128, 512, 1);
    ln_kernel<<<65536, 128, 0, stream>>>(edge_in, eout_f, eln0_g, eln0_b, edge1_f, edge1_bf, 128);
    gemm(edge1_bf, wm1_t, nullptr, eh2_bf, nullptr, nullptr, nullptr, 65536, 512, 128, 2);
    gemm(eh2_bf, wm2_t, em2_f, nullptr, b_em2, nullptr, nullptr, 65536, 128, 512, 1);
    ln_kernel<<<65536, 128, 0, stream>>>(edge1_f, em2_f, eln1_g, eln1_b, out_edge, nullptr, 128);
}
